// AttnBlockST_70007966925342
// MI455X (gfx1250) — matrix operand order repaired and checked by reading, not compiled, not run on hardware
//
#include <hip/hip_runtime.h>
#include <hip/hip_bf16.h>

// ---------------------------------------------------------------------------
// AttnBlockST for MI455X (gfx1250): bf16 WMMA GEMMs + f32 GN/softmax/residual
// ---------------------------------------------------------------------------

typedef __bf16 bf16_t;
typedef __attribute__((ext_vector_type(16))) __bf16 v16bf;
typedef __attribute__((ext_vector_type(8)))  __bf16 v8bf;
typedef __attribute__((ext_vector_type(8)))  float  v8f;

__device__ __forceinline__ bf16_t f2bf(float f) {
    unsigned u = __builtin_bit_cast(unsigned, f);
    u += 0x7fffu + ((u >> 16) & 1u);               // round-to-nearest-even
    unsigned short h = (unsigned short)(u >> 16);
    return __builtin_bit_cast(bf16_t, h);
}
__device__ __forceinline__ float bf2f(bf16_t b) {
    unsigned short h = __builtin_bit_cast(unsigned short, b);
    unsigned u = ((unsigned)h) << 16;
    return __builtin_bit_cast(float, u);
}

__device__ __forceinline__ v16bf cat8(v8bf lo, v8bf hi) {
    return __builtin_shufflevector(lo, hi, 0,1,2,3,4,5,6,7,8,9,10,11,12,13,14,15);
}

// A fragment 16x32 bf16 (M x K): lane<16 -> row=lane&15, K = {0..7, 16..23}
//                                lane>=16 ->             K = {8..15, 24..31}
__device__ __forceinline__ v16bf load_frag_a(const bf16_t* __restrict__ A, int lda,
                                             int row, int k0, int lane) {
    const int half = lane >> 4;
    const bf16_t* p = A + (size_t)row * lda + (size_t)(k0 + half * 8);
    v8bf lo = *(const v8bf*)p;           // 16B
    v8bf hi = *(const v8bf*)(p + 16);    // 16B
    return cat8(lo, hi);
}

// B fragment 32x16 bf16 (K x N), B stored column-contiguous (Bc[n*ldb + k]):
// lane<16 -> col = lane&15, K = {0..7, 16..23} ; lane>=16 -> K = {8..15, 24..31}
__device__ __forceinline__ v16bf load_frag_b(const bf16_t* __restrict__ Bc, int ldb,
                                             int col, int k0, int lane) {
    const int half = lane >> 4;
    const bf16_t* p = Bc + (size_t)col * ldb + (size_t)(k0 + half * 8);
    v8bf lo = *(const v8bf*)p;           // 16B
    v8bf hi = *(const v8bf*)(p + 16);    // 16B
    return cat8(lo, hi);
}

__device__ __forceinline__ v8f wmma_bf16(v16bf a, v16bf b, v8f c) {
    return __builtin_amdgcn_wmma_f32_16x16x32_bf16(false, a, false, b,
                                                   (short)0, c, false, false);
}

// ---------------------------------------------------------------------------
// Generic batched NT GEMM: D = scale * A(MxK) * B(KxN) + bias[m]  (bf16 in)
//   A row-major (lda=K-stride), B column-contiguous (ldb=K-stride per column)
//   block = 8 waves, wave = 32x32 macro-tile (2x2 WMMA tiles), block = 64x128
// store modes:
//   0: bf16  Db[n*ldd + m]                         (q/k/outh layouts, packed 16B)
//   1: bf16  Db[m*ldd + n]                         (logits / attn@V output)
//   2: bf16  Db[(n/Lsub)*M*Lsub + m*Lsub + n%Lsub] (v layout (batch, c, l))
//   3: f32   Df[a] = resid[a] + acc   (a as mode 2; spatial proj, in-place)
//   4: f32   final temporal proj: scatter to (b,c,t,h,w) + residual
// ---------------------------------------------------------------------------
__global__ __launch_bounds__(256)
void gemm_nt_wmma(const bf16_t* __restrict__ A, const bf16_t* __restrict__ B,
                  bf16_t* __restrict__ Db, float* __restrict__ Df,
                  const float* __restrict__ bias, const float* __restrict__ resid,
                  int M, int N, int K, int lda, int ldb, int ldd,
                  long long sA, long long sB, long long sD,
                  int mode, int Lsub, float scale)
{
    const int lane = threadIdx.x & 31;
    const int wid  = threadIdx.x >> 5;
    const int half = lane >> 4;
    const int l16  = lane & 15;

    const int m0 = blockIdx.y * 64  + (wid & 1) * 32;
    const int n0 = blockIdx.x * 128 + (wid >> 1) * 32;

    A += (size_t)blockIdx.z * (size_t)sA;
    B += (size_t)blockIdx.z * (size_t)sB;
    const size_t dOff = (size_t)blockIdx.z * (size_t)sD;

    v8f acc[2][2];
    #pragma unroll
    for (int i = 0; i < 2; ++i)
        #pragma unroll
        for (int j = 0; j < 2; ++j)
            acc[i][j] = (v8f){0.f,0.f,0.f,0.f,0.f,0.f,0.f,0.f};

    for (int k0 = 0; k0 < K; k0 += 32) {
        v16bf a0 = load_frag_a(A, lda, m0 + l16,      k0, lane);
        v16bf a1 = load_frag_a(A, lda, m0 + 16 + l16, k0, lane);
        v16bf b0 = load_frag_b(B, ldb, n0 + l16,      k0, lane);
        v16bf b1 = load_frag_b(B, ldb, n0 + 16 + l16, k0, lane);
        acc[0][0] = wmma_bf16(a0, b0, acc[0][0]);
        acc[0][1] = wmma_bf16(a0, b1, acc[0][1]);
        acc[1][0] = wmma_bf16(a1, b0, acc[1][0]);
        acc[1][1] = wmma_bf16(a1, b1, acc[1][1]);
    }

    #pragma unroll
    for (int i = 0; i < 2; ++i) {
        #pragma unroll
        for (int j = 0; j < 2; ++j) {
            const int n  = n0 + j * 16 + l16;
            const int mb = m0 + i * 16 + half * 8;
            float vals[8];
            #pragma unroll
            for (int r = 0; r < 8; ++r) {
                float v = acc[i][j][r] * scale;
                if (bias) v += bias[mb + r];
                vals[r] = v;
            }
            if (mode == 0) {
                v8bf o;
                #pragma unroll
                for (int r = 0; r < 8; ++r) o[r] = f2bf(vals[r]);
                *(v8bf*)(Db + dOff + (size_t)n * ldd + mb) = o;   // packed 16B
            } else if (mode == 1) {
                #pragma unroll
                for (int r = 0; r < 8; ++r)
                    Db[dOff + (size_t)(mb + r) * ldd + n] = f2bf(vals[r]);
            } else if (mode == 2) {
                #pragma unroll
                for (int r = 0; r < 8; ++r)
                    Db[dOff + (size_t)(n / Lsub) * (size_t)M * Lsub
                            + (size_t)(mb + r) * Lsub + (n % Lsub)] = f2bf(vals[r]);
            } else if (mode == 3) {
                #pragma unroll
                for (int r = 0; r < 8; ++r) {
                    size_t a = dOff + (size_t)(n / Lsub) * (size_t)M * Lsub
                                    + (size_t)(mb + r) * Lsub + (n % Lsub);
                    Df[a] = resid[a] + vals[r];
                }
            } else { // mode 4: n = (b*1024+hw)*16 + t, m = channel
                const int bhw = n >> 4, t = n & 15;
                const int bb = bhw >> 10, hw = bhw & 1023;
                #pragma unroll
                for (int r = 0; r < 8; ++r) {
                    const int c = mb + r;
                    size_t oa = (((size_t)(bb * 512 + c)) * 16 + t) * 1024 + hw;
                    size_t ra = (((size_t)(bb * 16 + t)) * 512 + c) * 1024 + hw;
                    Df[oa] = resid[ra] + vals[r];
                }
            }
        }
    }
}

// (b,c,t,h,w) f32 -> xs (bt, c, hw) f32, fully coalesced
__global__ void k_transpose_in(const float* __restrict__ x, float* __restrict__ xs) {
    size_t i = (size_t)blockIdx.x * 256 + threadIdx.x;   // 2*16*512*1024
    int hw = (int)(i & 1023);
    size_t r = i >> 10;
    int c  = (int)(r & 511);
    int bt = (int)(r >> 9);
    int t = bt & 15, b = bt >> 4;
    xs[i] = x[(((size_t)b * 512 + c) * 16 + t) * 1024 + hw];
}

__global__ void k_f32_to_bf16(const float* __restrict__ s, bf16_t* __restrict__ d, int n) {
    int i = blockIdx.x * 256 + threadIdx.x;
    if (i < n) d[i] = f2bf(s[i]);
}

// spatial GroupNorm: block = (group g, slice bt); stats over 16ch x 1024
// writes h in (bt, l, c) layout (bf16) for contiguous WMMA fragments
__global__ __launch_bounds__(256)
void k_gn_spatial(const float* __restrict__ xs, const float* __restrict__ gamma,
                  const float* __restrict__ beta, bf16_t* __restrict__ h) {
    const int g = blockIdx.x, bt = blockIdx.y;
    const float* base = xs + ((size_t)bt * 512 + g * 16) * 1024;
    __shared__ float s1[256], s2[256];
    float a = 0.f, b2 = 0.f;
    for (int i = threadIdx.x; i < 16 * 1024; i += 256) {
        float v = base[i];
        a += v; b2 += v * v;
    }
    s1[threadIdx.x] = a; s2[threadIdx.x] = b2;
    __syncthreads();
    for (int st = 128; st > 0; st >>= 1) {
        if (threadIdx.x < st) { s1[threadIdx.x] += s1[threadIdx.x + st];
                                s2[threadIdx.x] += s2[threadIdx.x + st]; }
        __syncthreads();
    }
    const float mean = s1[0] * (1.f / 16384.f);
    const float var  = s2[0] * (1.f / 16384.f) - mean * mean;
    const float rinv = rsqrtf(var + 1e-6f);
    for (int i = threadIdx.x; i < 16 * 1024; i += 256) {
        int c = g * 16 + (i >> 10), l = i & 1023;
        float y = (base[i] - mean) * rinv * gamma[c] + beta[c];
        h[((size_t)bt * 1024 + l) * 512 + c] = f2bf(y);
    }
}

// in-place row softmax over bf16 logits (row length 1024)
__global__ __launch_bounds__(256)
void k_softmax_rows(bf16_t* __restrict__ att) {
    bf16_t* row = att + ((size_t)blockIdx.y * 1024 + blockIdx.x) * 1024;
    __shared__ float red[256];
    float vals[4]; float mx = -1e30f;
    #pragma unroll
    for (int j = 0; j < 4; ++j) {
        vals[j] = bf2f(row[threadIdx.x * 4 + j]);
        mx = fmaxf(mx, vals[j]);
    }
    red[threadIdx.x] = mx; __syncthreads();
    for (int st = 128; st > 0; st >>= 1) {
        if (threadIdx.x < st) red[threadIdx.x] = fmaxf(red[threadIdx.x], red[threadIdx.x + st]);
        __syncthreads();
    }
    mx = red[0]; __syncthreads();
    float s = 0.f;
    #pragma unroll
    for (int j = 0; j < 4; ++j) { vals[j] = __expf(vals[j] - mx); s += vals[j]; }
    red[threadIdx.x] = s; __syncthreads();
    for (int st = 128; st > 0; st >>= 1) {
        if (threadIdx.x < st) red[threadIdx.x] += red[threadIdx.x + st];
        __syncthreads();
    }
    const float r = 1.f / red[0];
    #pragma unroll
    for (int j = 0; j < 4; ++j) row[threadIdx.x * 4 + j] = f2bf(vals[j] * r);
}

// temporal GroupNorm: block per (b, hw); stats over 16ch x 16t per group
// writes h in (bhw, t, c) layout (bf16)
__global__ __launch_bounds__(256)
void k_gn_temporal(const float* __restrict__ xs, const float* __restrict__ gamma,
                   const float* __restrict__ beta, bf16_t* __restrict__ h) {
    const int bhw = blockIdx.x;
    const int b = bhw >> 10, hw = bhw & 1023;
    const int lane = threadIdx.x & 31;
    __shared__ float gs[32], gs2[32];
    if (threadIdx.x < 32) { gs[threadIdx.x] = 0.f; gs2[threadIdx.x] = 0.f; }
    __syncthreads();
    float vals[32];
    #pragma unroll
    for (int g = 0; g < 32; ++g) {
        int e = g * 256 + threadIdx.x;
        int c = e >> 4, t = e & 15;
        float v = xs[(((size_t)(b * 16 + t)) * 512 + c) * 1024 + hw];
        vals[g] = v;
        float s = v, s2 = v * v;
        #pragma unroll
        for (int m = 16; m >= 1; m >>= 1) { s += __shfl_xor(s, m); s2 += __shfl_xor(s2, m); }
        if (lane == 0) { atomicAdd(&gs[g], s); atomicAdd(&gs2[g], s2); }
    }
    __syncthreads();
    #pragma unroll
    for (int g = 0; g < 32; ++g) {
        int e = g * 256 + threadIdx.x;
        int c = e >> 4, t = e & 15;
        float mean = gs[g] * (1.f / 256.f);
        float var  = gs2[g] * (1.f / 256.f) - mean * mean;
        float y = (vals[g] - mean) * rsqrtf(var + 1e-6f) * gamma[c] + beta[c];
        h[((size_t)bhw * 16 + t) * 512 + c] = f2bf(y);
    }
}

// fused temporal attention per (b,h,w): one wave per block
//   logits(16x16) = q^T k / sqrt(C) via WMMA K-loop; lane-shuffle softmax over
//   keys (the lane dim of the C/D layout); LDS transpose of the prob tile into
//   an A fragment (K padded 16->32); then 32 WMMAs over channels.
__global__ __launch_bounds__(32)
void k_attn_temporal(const bf16_t* __restrict__ q, const bf16_t* __restrict__ k,
                     const bf16_t* __restrict__ v, bf16_t* __restrict__ outh) {
    const int bhw  = blockIdx.x;
    const int lane = threadIdx.x & 31;
    const int half = lane >> 4, l16 = lane & 15;
    const bf16_t* qb = q + (size_t)bhw * 16 * 512;   // (t, c)
    const bf16_t* kb = k + (size_t)bhw * 16 * 512;   // (t, c)
    const bf16_t* vb = v + (size_t)bhw * 512 * 16;   // (c, t)

    v8f logit = (v8f){0.f,0.f,0.f,0.f,0.f,0.f,0.f,0.f};
    for (int k0 = 0; k0 < 512; k0 += 32) {
        v16bf a = load_frag_a(qb, 512, l16, k0, lane);
        v16bf b = load_frag_b(kb, 512, l16, k0, lane);
        logit = wmma_bf16(a, b, logit);
    }
    const float scale = 0.044194173824159216f;  // 512^-0.5
    // softmax across the 16 lanes of each half (key index = lane dim)
    float p[8];
    #pragma unroll
    for (int r = 0; r < 8; ++r) {
        float x = logit[r] * scale;
        float mx = x;
        #pragma unroll
        for (int m = 8; m >= 1; m >>= 1) mx = fmaxf(mx, __shfl_xor(mx, m));
        x = __expf(x - mx);
        float s = x;
        #pragma unroll
        for (int m = 8; m >= 1; m >>= 1) s += __shfl_xor(s, m);
        p[r] = x / s;
    }
    __shared__ float attnLds[16][17];
    #pragma unroll
    for (int r = 0; r < 8; ++r) attnLds[half * 8 + r][l16] = p[r];
    __syncthreads();
    // A fragment of attn (16x32, K=16..31 zero-padded)
    v16bf afrag;
    #pragma unroll
    for (int j = 0; j < 16; ++j) afrag[j] = f2bf(half == 0 ? attnLds[l16][j] : 0.f);

    for (int n0 = 0; n0 < 512; n0 += 16) {
        v16bf bfrag;
        if (half == 0) {                       // K = 0..15 (real data)
            const bf16_t* p2 = vb + (size_t)(n0 + l16) * 16;
            v8bf lo = *(const v8bf*)p2;
            v8bf hi = *(const v8bf*)(p2 + 8);
            bfrag = cat8(lo, hi);
        } else {                               // K = 16..31 padded with zero
            #pragma unroll
            for (int j = 0; j < 16; ++j) bfrag[j] = f2bf(0.f);
        }
        v8f o = (v8f){0.f,0.f,0.f,0.f,0.f,0.f,0.f,0.f};
        o = wmma_bf16(afrag, bfrag, o);
        #pragma unroll
        for (int r = 0; r < 8; ++r)
            outh[((size_t)bhw * 16 + half * 8 + r) * 512 + n0 + l16] = f2bf(o[r]);
    }
}

// ---------------------------------------------------------------------------
extern "C" void kernel_launch(void* const* d_in, const int* in_sizes, int n_in,
                              void* d_out, int out_size, void* d_ws, size_t ws_size,
                              hipStream_t stream) {
    const float* x       = (const float*)d_in[0];
    const float* gamma_s = (const float*)d_in[1];
    const float* beta_s  = (const float*)d_in[2];
    const float* wq_s    = (const float*)d_in[3];
    const float* bq_s    = (const float*)d_in[4];
    const float* wk_s    = (const float*)d_in[5];
    const float* bk_s    = (const float*)d_in[6];
    const float* wv_s    = (const float*)d_in[7];
    const float* bv_s    = (const float*)d_in[8];
    const float* wo_s    = (const float*)d_in[9];
    const float* bo_s    = (const float*)d_in[10];
    const float* gamma_t = (const float*)d_in[11];
    const float* beta_t  = (const float*)d_in[12];
    const float* wq_t    = (const float*)d_in[13];
    const float* bq_t    = (const float*)d_in[14];
    const float* wk_t    = (const float*)d_in[15];
    const float* bk_t    = (const float*)d_in[16];
    const float* wv_t    = (const float*)d_in[17];
    const float* bv_t    = (const float*)d_in[18];
    const float* wo_t    = (const float*)d_in[19];
    const float* bo_t    = (const float*)d_in[20];
    float* out = (float*)d_out;

    char* w = (char*)d_ws;
    float*  xs  = (float*) (w);                      // 64 MB (bt,c,l) f32, in-place residual
    bf16_t* h   = (bf16_t*)(w + (64ull  << 20));     // 32 MB normalized activations
    bf16_t* qb  = (bf16_t*)(w + (96ull  << 20));     // 32 MB (batch,l,c)
    bf16_t* kb  = (bf16_t*)(w + (128ull << 20));     // 32 MB (batch,l,c)
    bf16_t* vb  = (bf16_t*)(w + (160ull << 20));     // 32 MB (batch,c,l)
    bf16_t* att = (bf16_t*)(w + (192ull << 20));     // 64 MB logits->attn in place
    bf16_t* oh  = (bf16_t*)(w + (256ull << 20));     // 32 MB attn output (batch,l,c)
    bf16_t* wb  = (bf16_t*)(w + (288ull << 20));     // 4 MB: 8 bf16 weight matrices

    const int WN = 512 * 512;
    const float rs = 0.044194173824159216f;   // 1/sqrt(512)

    // weights -> bf16
    k_f32_to_bf16<<<WN / 256, 256, 0, stream>>>(wq_s, wb + 0 * (size_t)WN, WN);
    k_f32_to_bf16<<<WN / 256, 256, 0, stream>>>(wk_s, wb + 1 * (size_t)WN, WN);
    k_f32_to_bf16<<<WN / 256, 256, 0, stream>>>(wv_s, wb + 2 * (size_t)WN, WN);
    k_f32_to_bf16<<<WN / 256, 256, 0, stream>>>(wo_s, wb + 3 * (size_t)WN, WN);
    k_f32_to_bf16<<<WN / 256, 256, 0, stream>>>(wq_t, wb + 4 * (size_t)WN, WN);
    k_f32_to_bf16<<<WN / 256, 256, 0, stream>>>(wk_t, wb + 5 * (size_t)WN, WN);
    k_f32_to_bf16<<<WN / 256, 256, 0, stream>>>(wv_t, wb + 6 * (size_t)WN, WN);
    k_f32_to_bf16<<<WN / 256, 256, 0, stream>>>(wo_t, wb + 7 * (size_t)WN, WN);

    // x -> xs (bt, c, hw)
    k_transpose_in<<<16777216 / 256, 256, 0, stream>>>(x, xs);

    // ---- spatial stage ----
    k_gn_spatial<<<dim3(32, 32), 256, 0, stream>>>(xs, gamma_s, beta_s, h);

    // q, k: (o, l) stored n-major (bt,l,c); v stored (bt,c,l)
    gemm_nt_wmma<<<dim3(256, 8, 1), 256, 0, stream>>>(wb + 0 * (size_t)WN, h, qb, nullptr,
        bq_s, nullptr, 512, 32768, 512, 512, 512, 512, 0, 0, 0, /*mode*/0, 1, 1.f);
    gemm_nt_wmma<<<dim3(256, 8, 1), 256, 0, stream>>>(wb + 1 * (size_t)WN, h, kb, nullptr,
        bk_s, nullptr, 512, 32768, 512, 512, 512, 512, 0, 0, 0, /*mode*/0, 1, 1.f);
    gemm_nt_wmma<<<dim3(256, 8, 1), 256, 0, stream>>>(wb + 2 * (size_t)WN, h, vb, nullptr,
        bv_s, nullptr, 512, 32768, 512, 512, 512, 512, 0, 0, 0, /*mode*/2, 1024, 1.f);

    // logits[bt] = q^T k * rs : batched 1024x1024x512, row-major bf16
    gemm_nt_wmma<<<dim3(8, 16, 32), 256, 0, stream>>>(qb, kb, att, nullptr,
        nullptr, nullptr, 1024, 1024, 512, 512, 512, 1024,
        512LL * 1024, 512LL * 1024, 1024LL * 1024, /*mode*/1, 1, rs);

    k_softmax_rows<<<dim3(1024, 32), 256, 0, stream>>>(att);

    // attn @ v^T : batched 1024x512x1024 -> oh (bt, l, c)
    gemm_nt_wmma<<<dim3(4, 16, 32), 256, 0, stream>>>(att, vb, oh, nullptr,
        nullptr, nullptr, 1024, 512, 1024, 1024, 1024, 512,
        1024LL * 1024, 512LL * 1024, 512LL * 1024, /*mode*/1, 1, 1.f);

    // spatial proj + residual, in-place into xs (f32)
    gemm_nt_wmma<<<dim3(256, 8, 1), 256, 0, stream>>>(wb + 3 * (size_t)WN, oh, nullptr, xs,
        bo_s, xs, 512, 32768, 512, 512, 512, 512, 0, 0, 0, /*mode*/3, 1024, 1.f);

    // ---- temporal stage ----
    k_gn_temporal<<<2048, 256, 0, stream>>>(xs, gamma_t, beta_t, h);

    gemm_nt_wmma<<<dim3(256, 8, 1), 256, 0, stream>>>(wb + 4 * (size_t)WN, h, qb, nullptr,
        bq_t, nullptr, 512, 32768, 512, 512, 512, 512, 0, 0, 0, /*mode*/0, 1, 1.f);
    gemm_nt_wmma<<<dim3(256, 8, 1), 256, 0, stream>>>(wb + 5 * (size_t)WN, h, kb, nullptr,
        bk_t, nullptr, 512, 32768, 512, 512, 512, 512, 0, 0, 0, /*mode*/0, 1, 1.f);
    gemm_nt_wmma<<<dim3(256, 8, 1), 256, 0, stream>>>(wb + 6 * (size_t)WN, h, vb, nullptr,
        bv_t, nullptr, 512, 32768, 512, 512, 512, 512, 0, 0, 0, /*mode*/2, 16, 1.f);

    k_attn_temporal<<<2048, 32, 0, stream>>>(qb, kb, vb, oh);

    // temporal proj + residual, scattered to final (b,c,t,h,w) layout
    gemm_nt_wmma<<<dim3(256, 8, 1), 256, 0, stream>>>(wb + 7 * (size_t)WN, oh, nullptr, out,
        bo_t, xs, 512, 32768, 512, 512, 512, 512, 0, 0, 0, /*mode*/4, 1, 1.f);
}
